// Attention_63711544869380
// MI455X (gfx1250) — matrix operand order repaired and checked by reading, not compiled, not run on hardware
//
#include <hip/hip_runtime.h>

#define T_SEQ 4096
#define C_DIM 1024
#define H_HEADS 16
#define D_HEAD 64

typedef __bf16 bf16;
typedef __bf16 v16bf __attribute__((ext_vector_type(16)));
typedef __bf16 v8bf  __attribute__((ext_vector_type(8)));
typedef float  v8f   __attribute__((ext_vector_type(8)));

static __device__ __forceinline__ bf16 f2bf(float f) {
    unsigned u = __builtin_bit_cast(unsigned, f);
    u += 0x7FFFu + ((u >> 16) & 1u);            // round-to-nearest-even
    unsigned short h = (unsigned short)(u >> 16);
    return __builtin_bit_cast(bf16, h);
}

static __device__ __forceinline__ v16bf cat8(v8bf lo, v8bf hi) {
    return __builtin_shufflevector(lo, hi, 0,1,2,3,4,5,6,7,8,9,10,11,12,13,14,15);
}

static __device__ __forceinline__ v8f v8f_zero() {
    v8f z;
#pragma unroll
    for (int i = 0; i < 8; ++i) z[i] = 0.0f;
    return z;
}

static __device__ __forceinline__ v8f wmma_bf16(v16bf a, v16bf b, v8f c) {
    // D = A(16x32 bf16) * B(32x16 bf16) + C(16x16 f32)
    return __builtin_amdgcn_wmma_f32_16x16x32_bf16(
        false, a, false, b, (short)0, c, false, false);
}

// --- gfx1250 async global->LDS copy (ASYNCcnt-tracked), 16B per lane --------
static __device__ __forceinline__ void async_ld16(void* lds_ptr, const void* gptr) {
    unsigned loff = (unsigned)(size_t)lds_ptr;               // low 32b of generic = LDS offset
    unsigned long long ga = (unsigned long long)(size_t)gptr;
    asm volatile("global_load_async_to_lds_b128 %0, %1, off"
                 :: "v"(loff), "v"(ga) : "memory");
}
// wait until <=4 outstanding (tile in flight for next phase may remain)
static __device__ __forceinline__ void wait_async4() {
    asm volatile("s_wait_asynccnt 4" ::: "memory");
}
static __device__ __forceinline__ void wait_async0() {
    asm volatile("s_wait_asynccnt 0" ::: "memory");
}

// ---------------------------------------------------------------------------
// fp32 -> bf16 bulk convert, 8 elements/thread (one b128 store)
// ---------------------------------------------------------------------------
__global__ __launch_bounds__(256) void cvt_f32_bf16(
    const float* __restrict__ in, bf16* __restrict__ out, int n8)
{
    int i = blockIdx.x * blockDim.x + threadIdx.x;
    if (i >= n8) return;
    const float4* p = (const float4*)in + (size_t)i * 2;
    float4 a = p[0], b = p[1];
    v8bf o;
    o[0] = f2bf(a.x); o[1] = f2bf(a.y); o[2] = f2bf(a.z); o[3] = f2bf(a.w);
    o[4] = f2bf(b.x); o[5] = f2bf(b.y); o[6] = f2bf(b.z); o[7] = f2bf(b.w);
    ((v8bf*)out)[i] = o;
}

// ---------------------------------------------------------------------------
// GEMM (all-bf16 operands): Y[M=4096][N=1024] = X @ W^T + bias
// X[M][K], W[N][K] both bf16, K = 1024.
// MODE 0: fp32 out, flat (T,C)
// MODE 1: bf16 out, (H, T, D) layout   (for Q, K)
// MODE 2: bf16 out, (H, D, T) layout   (for V^T)
// 256 threads (8 waves), block tile 128x128, K-chunk 32.
// Double-buffered GLOBAL_LOAD_ASYNC_TO_LDS_B128 staging: next tile's DMA is
// in flight under the current tile's WMMAs; s_wait_asynccnt 4 relies on
// in-order async completion. Wave grid 4(M) x 2(N): 2x4 WMMA tiles/wave.
// ---------------------------------------------------------------------------
template <int MODE>
__global__ __launch_bounds__(256) void gemm_bf16_wmma(
    const bf16* __restrict__ X, const bf16* __restrict__ W,
    const float* __restrict__ bias, void* __restrict__ out)
{
    __shared__ bf16 lds_a0[128 * 32], lds_b0[128 * 32];
    __shared__ bf16 lds_a1[128 * 32], lds_b1[128 * 32];

    const int tid  = threadIdx.x;
    const int wid  = tid >> 5;
    const int lane = tid & 31;
    const int half = lane >> 4;
    const int l16  = lane & 15;
    const int wm   = wid & 3;       // 0..3 -> 32-row strip
    const int wn   = wid >> 2;      // 0..1 -> 64-col strip
    const int m0   = blockIdx.y * 128;
    const int n0   = blockIdx.x * 128;

    // tile staging: 128 rows x 32 cols bf16 = 512 chunks of 8 elems (16B);
    // thread covers chunks tid and tid+256
    const int r0 = tid >> 2,          c0 = (tid & 3) * 8;
    const int r1 = (tid + 256) >> 2,  c1 = ((tid + 256) & 3) * 8;

    v8f acc[2][4];
#pragma unroll
    for (int i = 0; i < 2; ++i)
#pragma unroll
        for (int j = 0; j < 4; ++j) acc[i][j] = v8f_zero();

    auto issue = [&](bf16* la, bf16* lb, int kk) {
        async_ld16(&la[r0 * 32 + c0], X + (size_t)(m0 + r0) * C_DIM + kk + c0);
        async_ld16(&la[r1 * 32 + c1], X + (size_t)(m0 + r1) * C_DIM + kk + c1);
        async_ld16(&lb[r0 * 32 + c0], W + (size_t)(n0 + r0) * C_DIM + kk + c0);
        async_ld16(&lb[r1 * 32 + c1], W + (size_t)(n0 + r1) * C_DIM + kk + c1);
    };
    auto compute = [&](const bf16* la, const bf16* lb) {
        // A operands: lane L<16 -> row L, K 0..15; L>=16 -> K 16..31 (contiguous)
        v16bf afrag[2];
#pragma unroll
        for (int mt = 0; mt < 2; ++mt) {
            int row = wm * 32 + mt * 16 + l16;
            afrag[mt] = cat8(*(const v8bf*)&la[row * 32 + half * 16],
                             *(const v8bf*)&la[row * 32 + half * 16 + 8]);
        }
        // B operands: lane L<16 -> col L, K 0..15; L>=16 -> K 16..31 (contiguous)
        v16bf bfrag[4];
#pragma unroll
        for (int nt = 0; nt < 4; ++nt) {
            int row = wn * 64 + nt * 16 + l16;
            bfrag[nt] = *(const v16bf*)&lb[row * 32 + half * 16];
        }
#pragma unroll
        for (int mt = 0; mt < 2; ++mt)
#pragma unroll
            for (int nt = 0; nt < 4; ++nt)
                acc[mt][nt] = wmma_bf16(afrag[mt], bfrag[nt], acc[mt][nt]);
    };

    issue(lds_a0, lds_b0, 0);                 // prologue: tile 0 -> buf0
    for (int k0 = 0; k0 < C_DIM; k0 += 64) {
        // phase A: tile k0 in buf0; overlap DMA of tile k0+32 into buf1
        issue(lds_a1, lds_b1, k0 + 32);       // k0+32 < C_DIM always (K multiple of 64)
        wait_async4();                        // tile k0 landed (in-order completion)
        __syncthreads();
        compute(lds_a0, lds_b0);
        __syncthreads();                      // all reads of buf0 done before reuse

        // phase B: tile k0+32 in buf1; overlap DMA of tile k0+64 into buf0
        if (k0 + 64 < C_DIM) {
            issue(lds_a0, lds_b0, k0 + 64);
            wait_async4();
        } else {
            wait_async0();                    // last tile: drain fully
        }
        __syncthreads();
        compute(lds_a1, lds_b1);
        __syncthreads();
    }

    // epilogue: C/D layout is col = lane%16, row = vgpr + 8*(lane/16)
#pragma unroll
    for (int nt = 0; nt < 4; ++nt) {
        int ncol = n0 + wn * 64 + nt * 16 + l16;
        float bv = bias[ncol];
#pragma unroll
        for (int mt = 0; mt < 2; ++mt)
#pragma unroll
            for (int r = 0; r < 8; ++r) {
                int mrow = m0 + wm * 32 + mt * 16 + r + 8 * half;
                float val = acc[mt][nt][r] + bv;
                if (MODE == 0) {
                    ((float*)out)[(size_t)mrow * C_DIM + ncol] = val;
                } else if (MODE == 1) {   // (H,T,D): h = ncol>>6, d = ncol&63
                    ((bf16*)out)[((size_t)(ncol >> 6) * T_SEQ + mrow) * D_HEAD + (ncol & 63)] = f2bf(val);
                } else {                  // (H,D,T): flat head*64+d == ncol
                    ((bf16*)out)[(size_t)ncol * T_SEQ + mrow] = f2bf(val);
                }
            }
    }
}

// ---------------------------------------------------------------------------
// Causal flash attention. grid = (T/64 q-blocks, H heads), 128 threads.
// Each wave owns a 16-row Q tile; online softmax over 32-key chunks.
// q,k: bf16 (H,T,64); vt: bf16 (H,64,T); y: bf16 (T,C).
// PBrelax's global-abs-max shift is uniform across a softmax row -> no-op;
// effective scale is 1/sqrt(D) = 0.125.
// ---------------------------------------------------------------------------
__global__ __launch_bounds__(128) void flash_attn_wmma(
    const bf16* __restrict__ q, const bf16* __restrict__ k,
    const bf16* __restrict__ vt, bf16* __restrict__ y)
{
    __shared__ bf16 lds_p[4][16][32];   // per-wave P tile (16 x 32 keys)

    const int tid  = threadIdx.x;
    const int wid  = tid >> 5;
    const int lane = tid & 31;
    const int half = lane >> 4;
    const int l16  = lane & 15;
    const int h    = blockIdx.y;
    const int qb   = blockIdx.x;
    const int q0w  = qb * 64 + wid * 16;     // first Q row of this wave
    const float scale = 0.125f;

    // Q tile as two K=32 A-operands (d 0..31 and 32..63), resident in VGPRs
    v16bf aq[2];
#pragma unroll
    for (int c = 0; c < 2; ++c) {
        const bf16* base = q + ((size_t)h * T_SEQ + q0w + l16) * D_HEAD + c * 32 + half * 16;
        aq[c] = cat8(*(const v8bf*)base, *(const v8bf*)(base + 8));
    }

    float m_s[8], l_s[8];
    v8f o[4];
#pragma unroll
    for (int r = 0; r < 8; ++r) { m_s[r] = -1e30f; l_s[r] = 0.0f; }
#pragma unroll
    for (int nt = 0; nt < 4; ++nt) o[nt] = v8f_zero();

    const int nkb = 2 * (qb + 1);            // uniform across the block
    for (int kb = 0; kb < nkb; ++kb) {
        const int key0 = kb * 32;

        // S = Q K^T for 32 keys: two 16-key column tiles, K(=d) chunks of 32
        v8f s0 = v8f_zero(), s1 = v8f_zero();
#pragma unroll
        for (int c = 0; c < 2; ++c) {
            const bf16* kb0 = k + ((size_t)h * T_SEQ + key0 + l16)      * D_HEAD + c * 32 + half * 16;
            const bf16* kb1 = k + ((size_t)h * T_SEQ + key0 + 16 + l16) * D_HEAD + c * 32 + half * 16;
            s0 = wmma_bf16(aq[c], *(const v16bf*)kb0, s0);
            s1 = wmma_bf16(aq[c], *(const v16bf*)kb1, s1);
        }

        // online softmax per row (row = r + 8*half; 16-lane halves hold one row each)
        float p0[8], p1[8];
#pragma unroll
        for (int r = 0; r < 8; ++r) {
            const int qg = q0w + r + 8 * half;
            float v0 = s0[r] * scale; if (key0 + l16      > qg) v0 = -1e30f;
            float v1 = s1[r] * scale; if (key0 + 16 + l16 > qg) v1 = -1e30f;
            float cm = fmaxf(v0, v1);
#pragma unroll
            for (int off = 1; off < 16; off <<= 1) cm = fmaxf(cm, __shfl_xor(cm, off, 16));
            const float mn   = fmaxf(m_s[r], cm);
            const float corr = __expf(m_s[r] - mn);
            const float e0 = __expf(v0 - mn);
            const float e1 = __expf(v1 - mn);
            float rs = e0 + e1;
#pragma unroll
            for (int off = 1; off < 16; off <<= 1) rs += __shfl_xor(rs, off, 16);
            l_s[r] = l_s[r] * corr + rs;
            m_s[r] = mn;
#pragma unroll
            for (int nt = 0; nt < 4; ++nt) o[nt][r] *= corr;
            p0[r] = e0; p1[r] = e1;
        }

        // C/D-layout -> A-layout conversion for P via per-wave LDS tile
        __syncthreads();
#pragma unroll
        for (int r = 0; r < 8; ++r) {
            int mrow = r + 8 * half;
            lds_p[wid][mrow][l16]      = f2bf(p0[r]);
            lds_p[wid][mrow][16 + l16] = f2bf(p1[r]);
        }
        __syncthreads();
        v16bf pa = cat8(*(const v8bf*)&lds_p[wid][l16][half * 16],
                        *(const v8bf*)&lds_p[wid][l16][half * 16 + 8]);

        // O += P @ V : V^T rows give contiguous 32B B-operand loads
#pragma unroll
        for (int nt = 0; nt < 4; ++nt) {
            const bf16* vb = vt + ((size_t)h * D_HEAD + nt * 16 + l16) * T_SEQ + key0 + half * 16;
            o[nt] = wmma_bf16(pa, *(const v16bf*)vb, o[nt]);
        }
    }

    // normalize and write (T, C) bf16, head-merged (feeds final bf16 GEMM)
#pragma unroll
    for (int nt = 0; nt < 4; ++nt)
#pragma unroll
        for (int r = 0; r < 8; ++r) {
            int qrow = q0w + r + 8 * half;
            int dcol = nt * 16 + l16;
            y[(size_t)qrow * C_DIM + h * D_HEAD + dcol] = f2bf(o[nt][r] / l_s[r]);
        }
}

// ---------------------------------------------------------------------------
extern "C" void kernel_launch(void* const* d_in, const int* in_sizes, int n_in,
                              void* d_out, int out_size, void* d_ws, size_t ws_size,
                              hipStream_t stream) {
    (void)in_sizes; (void)n_in; (void)out_size; (void)ws_size;
    const float* query = (const float*)d_in[0];
    const float* key_  = (const float*)d_in[1];
    const float* value = (const float*)d_in[2];
    // d_in[3] = att_mask: causal tril, baked into the flash kernel
    const float* Wq = (const float*)d_in[4];
    const float* bq = (const float*)d_in[5];
    const float* Wk = (const float*)d_in[6];
    const float* bk = (const float*)d_in[7];
    const float* Wv = (const float*)d_in[8];
    const float* bv = (const float*)d_in[9];
    const float* Wp = (const float*)d_in[10];
    const float* bp = (const float*)d_in[11];

    char* ws = (char*)d_ws;
    const size_t tc  = (size_t)T_SEQ * C_DIM;   // 4M elems
    const size_t cc  = (size_t)C_DIM * C_DIM;   // 1M elems
    bf16* xq  = (bf16*)(ws);                    // bf16 copies of activations
    bf16* xk  = xq  + tc;
    bf16* xv  = xk  + tc;
    bf16* wqb = xv  + tc;                       // bf16 copies of weights
    bf16* wkb = wqb + cc;
    bf16* wvb = wkb + cc;
    bf16* wpb = wvb + cc;
    bf16* q_bf  = wpb  + cc;                    // (H,T,D)
    bf16* k_bf  = q_bf + tc;                    // (H,T,D)
    bf16* vt_bf = k_bf + tc;                    // (H,D,T)
    bf16* y_bf  = vt_bf + tc;                   // (T,C)

    const int tc8 = (int)(tc / 8), cc8 = (int)(cc / 8);
    cvt_f32_bf16<<<tc8 / 256, 256, 0, stream>>>(query, xq,  tc8);
    cvt_f32_bf16<<<tc8 / 256, 256, 0, stream>>>(key_,  xk,  tc8);
    cvt_f32_bf16<<<tc8 / 256, 256, 0, stream>>>(value, xv,  tc8);
    cvt_f32_bf16<<<cc8 / 256, 256, 0, stream>>>(Wq,    wqb, cc8);
    cvt_f32_bf16<<<cc8 / 256, 256, 0, stream>>>(Wk,    wkb, cc8);
    cvt_f32_bf16<<<cc8 / 256, 256, 0, stream>>>(Wv,    wvb, cc8);
    cvt_f32_bf16<<<cc8 / 256, 256, 0, stream>>>(Wp,    wpb, cc8);

    dim3 gg(C_DIM / 128, T_SEQ / 128);   // (N tiles, M tiles)
    gemm_bf16_wmma<1><<<gg, 256, 0, stream>>>(xq, wqb, bq, q_bf);
    gemm_bf16_wmma<1><<<gg, 256, 0, stream>>>(xk, wkb, bk, k_bf);
    gemm_bf16_wmma<2><<<gg, 256, 0, stream>>>(xv, wvb, bv, vt_bf);

    flash_attn_wmma<<<dim3(T_SEQ / 64, H_HEADS), 128, 0, stream>>>(q_bf, k_bf, vt_bf, y_bf);

    gemm_bf16_wmma<0><<<gg, 256, 0, stream>>>(y_bf, wpb, bp, (float*)d_out);
}
